// BiLevelRoutingAttention_4045859193028
// MI455X (gfx1250) — matrix operand order repaired and checked by reading, not compiled, not run on hardware
//
#include <hip/hip_runtime.h>

typedef __attribute__((ext_vector_type(16))) _Float16 v16h;
typedef __attribute__((ext_vector_type(8)))  _Float16 v8h;
typedef __attribute__((ext_vector_type(4)))  _Float16 v4h;
typedef __attribute__((ext_vector_type(8)))  float    v8f;
typedef __attribute__((ext_vector_type(4)))  unsigned int v4u;
typedef __attribute__((ext_vector_type(8)))  int      v8i;
typedef __attribute__((ext_vector_type(4)))  int      v4i;

#define T_    4
#define B_    2
#define W_    32
#define S_    256
#define C_    256
#define H_    8
#define D_    32
#define K3_   768
#define TOPK_ 4
#define NTOK  (T_*B_*W_*S_)   // 65536 tokens

#if defined(__gfx1250__) && __has_builtin(__builtin_amdgcn_tensor_load_to_lds) && \
    __has_builtin(__builtin_amdgcn_s_wait_tensorcnt)
#define USE_TDM 1
#else
#define USE_TDM 0
#endif

#if USE_TDM
// Issue one TDM 2D tile load: [tile_dim1=256 rows][tile_dim0=32 elems] of
// 2-byte elements from a row-major tensor with stride0=256 into LDS
// (rows stored back-to-back, 64B apart -> matches [c][32] chunk layout).
__device__ __forceinline__ void tdm_load_tile_2d(unsigned lds_addr,
                                                 unsigned long long gaddr) {
  v4u g0;
  g0[0] = 1u;                                        // count=1, user mode
  g0[1] = lds_addr;                                  // D#.lds_addr (bytes)
  g0[2] = (unsigned)(gaddr & 0xFFFFFFFFull);         // global_addr[31:0]
  g0[3] = (unsigned)((gaddr >> 32) & 0x1FFFFFFull)   // global_addr[56:32]
        | (2u << 30);                                // type = 2 ("image")
  v8i g1;
  g1[0] = 0x10000;                                   // data_size=1 (2 bytes)
  g1[1] = (int)(256u << 16);                         // tensor_dim0 = 256
  g1[2] = (int)(256u << 16);                         // tensor_dim1 = 256
  g1[3] = (int)(32u  << 16);                         // tile_dim0  = 32
  g1[4] = 256;                                       // tile_dim1=256, tile_dim2=0
  g1[5] = 256;                                       // tensor_dim0_stride = 256
  g1[6] = 0;
  g1[7] = 0;
  v4i z4 = {};
#if __clang_major__ >= 23
  v8i z8 = {};
  __builtin_amdgcn_tensor_load_to_lds(g0, g1, z4, z4, z8, 0);
#else
  __builtin_amdgcn_tensor_load_to_lds(g0, g1, z4, z4, 0);
#endif
}
#endif

// ---------------------------------------------------------------------------
// Kernel 1: region means  region[b,w,c] = mean over (t, s) of window tokens
// ---------------------------------------------------------------------------
__global__ __launch_bounds__(256) void region_mean_kernel(
    const float* __restrict__ x, float* __restrict__ region) {
  int bw = blockIdx.x;            // 0..63  (b*W + w)
  int b  = bw >> 5, w = bw & 31;
  int c  = threadIdx.x;
  int wt = w >> 4, wh = (w >> 2) & 3, ww = w & 3;
  float acc = 0.f;
  for (int t = 0; t < T_; ++t) {
    for (int s = 0; s < S_; ++s) {
      int st = s >> 6, sh = (s >> 3) & 7, sw = s & 7;
      int Lti = wt*4 + st, Lhi = wh*8 + sh, Lwi = ww*8 + sw;
      size_t off = ((((size_t)(t*B_ + b)*8 + Lti)*32 + Lhi)*32 + Lwi)*C_ + c;
      acc += x[off];
    }
  }
  region[bw*C_ + c] = acc * (1.0f/1024.0f);
}

// ---------------------------------------------------------------------------
// Kernel 2: routing scores + top-4 (ties -> lowest index, matches lax.top_k)
// ---------------------------------------------------------------------------
__global__ __launch_bounds__(32) void route_topk_kernel(
    const float* __restrict__ region, int* __restrict__ idx) {
  int bw = blockIdx.x;
  int b  = bw >> 5, w = bw & 31;
  int v  = threadIdx.x;           // 0..31
  const float* rw = region + (size_t)(b*W_ + w)*C_;
  const float* rv = region + (size_t)(b*W_ + v)*C_;
  float s = 0.f;
  for (int c = 0; c < C_; ++c) s += rw[c]*rv[c];
  s *= 0.17677669529663687f;      // D^-0.5, D=32
  __shared__ float sc[W_];
  sc[v] = s;
  __syncthreads();
  if (v == 0) {
    for (int j = 0; j < TOPK_; ++j) {
      float best = -3.4e38f; int bi = 0;
      for (int u = 0; u < W_; ++u)
        if (sc[u] > best) { best = sc[u]; bi = u; }
      sc[bi] = -3.4e38f;
      idx[bw*TOPK_ + j] = bi;
    }
  }
}

// ---------------------------------------------------------------------------
// Kernel 3: QKV projection (WMMA f16, f32 acc) + LIF spikes.
//   Y = Xw @ Wqkv + b ; spike = (Y >= 2.0)
//   q -> row-major [token][C] f16; k,v -> transposed [win][C][S] f16
// ---------------------------------------------------------------------------
__global__ __launch_bounds__(256) void qkv_lif_kernel(
    const float* __restrict__ x, const float* __restrict__ w_qkv,
    const float* __restrict__ b_qkv,
    _Float16* __restrict__ q_rm, _Float16* __restrict__ k_t,
    _Float16* __restrict__ v_t) {
  __shared__ __attribute__((aligned(32))) _Float16 ldsA[128*32]; // [m][k]
  __shared__ __attribute__((aligned(32))) _Float16 ldsB[128*32]; // [n][k]
  int tid  = threadIdx.x;
  int nblk = blockIdx.x;          // 0..5
  int mblk = blockIdx.y;          // 0..511

  // Precompute gathered x row base offsets (window partition) for staging.
  size_t xoff[4]; int arow[4];
  for (int p = 0; p < 4; ++p) {
    int i = p*32 + (tid >> 3);
    arow[p] = i;
    int token = mblk*128 + i;
    int s = token & 255, wl = (token >> 8) & 31, tb = token >> 13;
    int wt = wl >> 4, wh = (wl >> 2) & 3, ww = wl & 3;
    int st = s >> 6,  sh = (s >> 3) & 7,  sw = s & 7;
    int Lti = wt*4 + st, Lhi = wh*8 + sh, Lwi = ww*8 + sw;
    xoff[p] = ((((size_t)tb*8 + Lti)*32 + Lhi)*32 + Lwi)*(size_t)C_;
  }

  int wv = tid >> 5, lane = tid & 31;
  int mwave = wv >> 1, nwave = wv & 1;
  int lm = lane & 15, khi = lane >> 4;

  v8f acc[2][4];
  {
    v8f z = {};
    for (int a = 0; a < 2; ++a)
      for (int n = 0; n < 4; ++n) acc[a][n] = z;
  }

  for (int kc = 0; kc < 8; ++kc) {
    // Stage A tile: 128 gathered rows x 32 k (f32 -> f16).
    for (int p = 0; p < 4; ++p) {
      const float* src = x + xoff[p] + kc*32 + (tid & 7)*4;
      float4 f = *(const float4*)src;
      if (kc < 7) __builtin_prefetch(src + 32, 0, 1);  // global_prefetch_b8
      v4h h4;
      h4[0] = (_Float16)f.x; h4[1] = (_Float16)f.y;
      h4[2] = (_Float16)f.z; h4[3] = (_Float16)f.w;
      *(v4h*)&ldsA[arow[p]*32 + (tid & 7)*4] = h4;
    }
    // Stage B tile transposed: w_qkv[k][n] -> ldsB[n][k].
    for (int p = 0; p < 4; ++p) {
      int kk = p*8 + (tid >> 5);
      int n0 = (tid & 31)*4;
      float4 f = *(const float4*)(w_qkv + (size_t)(kc*32 + kk)*K3_ + nblk*128 + n0);
      ldsB[(n0+0)*32 + kk] = (_Float16)f.x;
      ldsB[(n0+1)*32 + kk] = (_Float16)f.y;
      ldsB[(n0+2)*32 + kk] = (_Float16)f.z;
      ldsB[(n0+3)*32 + kk] = (_Float16)f.w;
    }
    __syncthreads();

    v16h afr[2];
    for (int mi = 0; mi < 2; ++mi) {
      const _Float16* base = &ldsA[(mwave*32 + mi*16 + lm)*32];
      union { v16h v; v8h h[2]; } u;
      u.h[0] = *(const v8h*)(base + khi*8);
      u.h[1] = *(const v8h*)(base + 16 + khi*8);
      afr[mi] = u.v;
    }
    for (int ni = 0; ni < 4; ++ni) {
      const _Float16* bb = &ldsB[(nwave*64 + ni*16 + lm)*32];
      union { v16h v; v8h h[2]; } ub;
      ub.h[0] = *(const v8h*)(bb + khi*8);
      ub.h[1] = *(const v8h*)(bb + 16 + khi*8);
      v16h bfr = ub.v;
      for (int mi = 0; mi < 2; ++mi)
        acc[mi][ni] = __builtin_amdgcn_wmma_f32_16x16x32_f16(
            false, afr[mi], false, bfr, (short)0, acc[mi][ni], false, false);
    }
    __syncthreads();
  }

  // Epilogue: bias + LIF threshold, scatter spikes.
  // For k/v (transposed layout) the 8 accumulator rows map to 8 consecutive
  // s positions at fixed channel -> pack into one 16B store.
  for (int mi = 0; mi < 2; ++mi)
    for (int ni = 0; ni < 4; ++ni) {
      int nglob = nblk*128 + nwave*64 + ni*16 + lm;  // 0..767
      float bias = b_qkv[nglob];
      int which = nglob >> 8, c = nglob & 255;
      int m0 = mwave*32 + mi*16 + khi*8;
      int token0 = mblk*128 + m0;
      if (which == 0) {
        for (int r = 0; r < 8; ++r) {
          float y = acc[mi][ni][r] + bias;
          q_rm[(size_t)(token0 + r)*C_ + c] =
              (y >= 2.0f) ? (_Float16)1.0f : (_Float16)0.0f;
        }
      } else {
        v8h pack;
        for (int r = 0; r < 8; ++r) {
          float y = acc[mi][ni][r] + bias;
          pack[r] = (y >= 2.0f) ? (_Float16)1.0f : (_Float16)0.0f;
        }
        int wl = token0 >> 8, s0 = token0 & 255;   // s0 multiple of 8
        _Float16* dst = (which == 1) ? k_t : v_t;
        *(v8h*)(dst + (size_t)wl*(C_*S_) + (size_t)c*S_ + s0) = pack;
      }
    }
}

// ---------------------------------------------------------------------------
// Kernel 4: routed linear attention per (t,b,w), one wave per head.
//   kv = k_g^T v_g (f16 WMMA over gathered 1024 tokens), den = q . sum(k_g),
//   out = (q @ kv) / (den + 1e-6) -> attn f16 [token][C]
//   k/v chunks stream through the Tensor Data Mover, double-buffered so the
//   DMA for chunk i+1 overlaps the WMMAs on chunk i (TENSORcnt <= 2 wait).
//   Phase-2 arrays (kv, ksum, den) alias the streaming buffers (64KB total).
//   Buffer addresses are computed by index arithmetic (no pointer arrays:
//   LDS addrspacecast constants in static initializers break ld.lld).
// ---------------------------------------------------------------------------
#define KBUF(n) (smem_h + (n)*8192)            // k tiles: halves [0, 16384)
#define VBUF(n) (smem_h + 16384 + (n)*8192)    // v tiles: halves [16384, 32768)

__global__ __launch_bounds__(256) void attn_kernel(
    const _Float16* __restrict__ q_rm, const _Float16* __restrict__ k_t,
    const _Float16* __restrict__ v_t, const int* __restrict__ idx,
    _Float16* __restrict__ attn) {
  __shared__ __attribute__((aligned(32))) _Float16 smem_h[32768]; // 64 KiB

  int tid = threadIdx.x;
  int h = tid >> 5, lane = tid & 31;
  int lm = lane & 15, khi = lane >> 4;
  int bx = blockIdx.x;            // 0..255 : (t*B+b)*W + w
  int w = bx & 31, tb = bx >> 5;
  int b = tb & 1;

  int gw[TOPK_];
  for (int j = 0; j < TOPK_; ++j)
    gw[j] = tb*W_ + idx[(b*W_ + w)*TOPK_ + j];

#if USE_TDM
  auto issue_pair = [&](int nit) {
    int jj = nit >> 3, ss = (nit & 7)*32;
    int bufn = nit & 1;
    const _Float16* ks = k_t + (size_t)gw[jj]*(C_*S_) + ss;
    const _Float16* vs = v_t + (size_t)gw[jj]*(C_*S_) + ss;
    tdm_load_tile_2d((unsigned)(size_t)KBUF(bufn), (unsigned long long)(size_t)ks);
    tdm_load_tile_2d((unsigned)(size_t)VBUF(bufn), (unsigned long long)(size_t)vs);
  };
  if (h == 0) issue_pair(0);      // prologue: start chunk 0
#endif

  v8f kvacc[2][2];
  { v8f z = {}; for (int i=0;i<2;++i) for (int j=0;j<2;++j) kvacc[i][j] = z; }
  float kacc = 0.f;

  for (int it = 0; it < 32; ++it) {
    int cur = it & 1;
    _Float16* kc = KBUF(cur);
    _Float16* vc = VBUF(cur);
#if USE_TDM
    if (h == 0) {
      if (it < 31) {
        issue_pair(it + 1);                       // prefetch next chunk
        __builtin_amdgcn_s_wait_tensorcnt(2);     // current pair has landed
      } else {
        __builtin_amdgcn_s_wait_tensorcnt(0);
      }
    }
    __syncthreads();
#else
    {
      int j = it >> 3, s0 = (it & 7)*32;
      const _Float16* ks = k_t + (size_t)gw[j]*(C_*S_) + (size_t)tid*S_ + s0;
      const _Float16* vs = v_t + (size_t)gw[j]*(C_*S_) + (size_t)tid*S_ + s0;
      *(v16h*)&kc[tid*32]      = *(const v16h*)ks;
      *(v16h*)&kc[tid*32 + 16] = *(const v16h*)(ks + 16);
      *(v16h*)&vc[tid*32]      = *(const v16h*)vs;
      *(v16h*)&vc[tid*32 + 16] = *(const v16h*)(vs + 16);
    }
    __syncthreads();
#endif

    v16h af[2], bf[2];
    for (int dt = 0; dt < 2; ++dt) {                 // A = k^T rows d
      const _Float16* base = kc + (h*32 + dt*16 + lm)*32;
      union { v16h v; v8h hh[2]; } u;
      u.hh[0] = *(const v8h*)(base + khi*8);
      u.hh[1] = *(const v8h*)(base + 16 + khi*8);
      af[dt] = u.v;
    }
    for (int et = 0; et < 2; ++et)                   // B = v cols e
    {
      const _Float16* bb = vc + (h*32 + et*16 + lm)*32;
      union { v16h v; v8h hh[2]; } ub;
      ub.hh[0] = *(const v8h*)(bb + khi*8);
      ub.hh[1] = *(const v8h*)(bb + 16 + khi*8);
      bf[et] = ub.v;
    }

    for (int dt = 0; dt < 2; ++dt)
      for (int et = 0; et < 2; ++et)
        kvacc[dt][et] = __builtin_amdgcn_wmma_f32_16x16x32_f16(
            false, af[dt], false, bf[et], (short)0, kvacc[dt][et], false, false);

    // ksum: lane = d, sum spikes over the 32 staged tokens.
    {
      const _Float16* kr = kc + (h*32 + lane)*32;
      for (int n = 0; n < 32; ++n) kacc += (float)kr[n];
    }
    __syncthreads();   // all reads done before next DMA / aliased phase-2
  }

  // Phase 2 aliases the streaming buffers (ordered by the loop-exit barrier).
  _Float16* kvl   = smem_h;                        // [h][e][d], 16 KiB
  float*    ksuml = (float*)(smem_h + 8192);       // [h][d], 1 KiB
  float*    denl  = (float*)(smem_h + 8192 + 512); // [h][s], 8 KiB

  // Publish kv (as f16, exact: integer counts <= 1024) and ksum.
  for (int dt = 0; dt < 2; ++dt)
    for (int et = 0; et < 2; ++et)
      for (int r = 0; r < 8; ++r) {
        int d = dt*16 + khi*8 + r;
        int e = et*16 + lm;
        kvl[(h*32 + e)*32 + d] = (_Float16)kvacc[dt][et][r];
      }
  ksuml[h*32 + lane] = kacc;
  __syncthreads();

  int tok0 = bx * S_;
  // Denominator: den[s] = q[s,:] . ksum
  for (int i = 0; i < 8; ++i) {
    int s = i*32 + lane;
    const _Float16* qr = q_rm + (size_t)(tok0 + s)*C_ + h*32;
    float dacc = 0.f;
    for (int d = 0; d < 32; ++d) dacc += (float)qr[d] * ksuml[h*32 + d];
    denl[h*S_ + s] = dacc;
  }
  __syncthreads();

  // out = q @ kv, normalize, store attn f16.
  for (int mt = 0; mt < 16; ++mt) {
    const _Float16* qr = q_rm + (size_t)(tok0 + mt*16 + lm)*C_ + h*32;
    union { v16h v; v8h hh[2]; } u;
    u.hh[0] = *(const v8h*)(qr + khi*8);
    u.hh[1] = *(const v8h*)(qr + 16 + khi*8);
    v16h aq = u.v;
    for (int et = 0; et < 2; ++et) {
      const _Float16* bb = kvl + (h*32 + et*16 + lm)*32;
      union { v16h v; v8h hh[2]; } ub;
      ub.hh[0] = *(const v8h*)(bb + khi*8);
      ub.hh[1] = *(const v8h*)(bb + 16 + khi*8);
      v16h bkv = ub.v;
      v8f o = {};
      o = __builtin_amdgcn_wmma_f32_16x16x32_f16(
          false, aq, false, bkv, (short)0, o, false, false);
      for (int r = 0; r < 8; ++r) {
        int s_out = mt*16 + khi*8 + r;
        int e = et*16 + lm;
        float val = o[r] / (denl[h*S_ + s_out] + 1e-6f);
        attn[(size_t)(tok0 + s_out)*C_ + h*32 + e] = (_Float16)val;
      }
    }
  }
}

// ---------------------------------------------------------------------------
// Kernel 5: output projection (WMMA) + bias + window reverse scatter (f32).
// ---------------------------------------------------------------------------
__global__ __launch_bounds__(256) void proj_kernel(
    const _Float16* __restrict__ attn, const float* __restrict__ w_proj,
    const float* __restrict__ b_proj, float* __restrict__ out) {
  __shared__ __attribute__((aligned(32))) _Float16 ldsA[128*32];
  __shared__ __attribute__((aligned(32))) _Float16 ldsB[128*32];
  int tid  = threadIdx.x;
  int nblk = blockIdx.x;          // 0..1
  int mblk = blockIdx.y;          // 0..511
  int wv = tid >> 5, lane = tid & 31;
  int mwave = wv >> 1, nwave = wv & 1;
  int lm = lane & 15, khi = lane >> 4;

  v8f acc[2][4];
  { v8f z = {}; for (int a=0;a<2;++a) for (int n=0;n<4;++n) acc[a][n] = z; }

  for (int kc = 0; kc < 8; ++kc) {
    for (int p = 0; p < 4; ++p) {                    // A: attn rows, already f16
      int i = p*32 + (tid >> 3);
      int token = mblk*128 + i;
      const _Float16* src = attn + (size_t)token*C_ + kc*32 + (tid & 7)*4;
      if (kc < 7) __builtin_prefetch(src + 32, 0, 1);
      *(v4h*)&ldsA[i*32 + (tid & 7)*4] = *(const v4h*)src;
    }
    for (int p = 0; p < 4; ++p) {                    // B: w_proj -> [n][k]
      int kk = p*8 + (tid >> 5);
      int n0 = (tid & 31)*4;
      float4 f = *(const float4*)(w_proj + (size_t)(kc*32 + kk)*C_ + nblk*128 + n0);
      ldsB[(n0+0)*32 + kk] = (_Float16)f.x;
      ldsB[(n0+1)*32 + kk] = (_Float16)f.y;
      ldsB[(n0+2)*32 + kk] = (_Float16)f.z;
      ldsB[(n0+3)*32 + kk] = (_Float16)f.w;
    }
    __syncthreads();

    v16h afr[2];
    for (int mi = 0; mi < 2; ++mi) {
      const _Float16* base = &ldsA[(mwave*32 + mi*16 + lm)*32];
      union { v16h v; v8h h[2]; } u;
      u.h[0] = *(const v8h*)(base + khi*8);
      u.h[1] = *(const v8h*)(base + 16 + khi*8);
      afr[mi] = u.v;
    }
    for (int ni = 0; ni < 4; ++ni) {
      const _Float16* bb = &ldsB[(nwave*64 + ni*16 + lm)*32];
      union { v16h v; v8h h[2]; } ub;
      ub.h[0] = *(const v8h*)(bb + khi*8);
      ub.h[1] = *(const v8h*)(bb + 16 + khi*8);
      v16h bfr = ub.v;
      for (int mi = 0; mi < 2; ++mi)
        acc[mi][ni] = __builtin_amdgcn_wmma_f32_16x16x32_f16(
            false, afr[mi], false, bfr, (short)0, acc[mi][ni], false, false);
    }
    __syncthreads();
  }

  for (int mi = 0; mi < 2; ++mi)
    for (int ni = 0; ni < 4; ++ni) {
      int c = nblk*128 + nwave*64 + ni*16 + lm;
      float bias = b_proj[c];
      for (int r = 0; r < 8; ++r) {
        int m = mwave*32 + mi*16 + khi*8 + r;
        int token = mblk*128 + m;
        int s = token & 255, wl = (token >> 8) & 31, tb = token >> 13;
        int wt = wl >> 4, wh = (wl >> 2) & 3, ww = wl & 3;
        int st = s >> 6,  sh = (s >> 3) & 7,  sw = s & 7;
        int Lti = wt*4 + st, Lhi = wh*8 + sh, Lwi = ww*8 + sw;
        size_t off = ((((size_t)tb*8 + Lti)*32 + Lhi)*32 + Lwi)*C_ + c;
        out[off] = acc[mi][ni][r] + bias;
      }
    }
}

// ---------------------------------------------------------------------------
extern "C" void kernel_launch(void* const* d_in, const int* in_sizes, int n_in,
                              void* d_out, int out_size, void* d_ws, size_t ws_size,
                              hipStream_t stream) {
  const float* x      = (const float*)d_in[0];
  const float* w_qkv  = (const float*)d_in[1];
  const float* b_qkv  = (const float*)d_in[2];
  const float* w_proj = (const float*)d_in[3];
  const float* b_proj = (const float*)d_in[4];
  float* out = (float*)d_out;

  char* ws = (char*)d_ws;
  const size_t SZ_SPK = (size_t)NTOK * C_ * sizeof(_Float16);   // 32 MiB
  float*    region = (float*)(ws);
  int*      idx    = (int*)(ws + 64*1024);
  _Float16* q_rm   = (_Float16*)(ws + 128*1024);
  _Float16* k_t    = (_Float16*)(ws + 128*1024 + SZ_SPK);
  _Float16* v_t    = (_Float16*)(ws + 128*1024 + 2*SZ_SPK);
  _Float16* attn   = (_Float16*)(ws + 128*1024 + 3*SZ_SPK);

  region_mean_kernel<<<B_*W_, 256, 0, stream>>>(x, region);
  route_topk_kernel <<<B_*W_, 32, 0, stream>>>(region, idx);
  qkv_lif_kernel    <<<dim3(6, 512), 256, 0, stream>>>(x, w_qkv, b_qkv, q_rm, k_t, v_t);
  attn_kernel       <<<T_*B_*W_, 256, 0, stream>>>(q_rm, k_t, v_t, idx, attn);
  proj_kernel       <<<dim3(2, 512), 256, 0, stream>>>(attn, w_proj, b_proj, out);
}
